// Bionic_68092411510857
// MI455X (gfx1250) — matrix operand order repaired and checked by reading, not compiled, not run on hardware
//
#include <hip/hip_runtime.h>
#include <hip/hip_bf16.h>

// ---------------- static problem sizes (match reference) ----------------
#define IN_SIZE   50000
#define MMOD      3
#define N0S       40960
#define N1S       16384
#define BATCH     4096
#define HEADS     4
#define DIMC      64
#define HD        256
#define DEG       16
#define E0S       (N1S*DEG)
#define E1S       (BATCH*DEG)
#define NEG_SLOPE 0.2f

#define TN        64      // block column tile (4 x 16-wide WMMA tiles)
#define GWAVES    8       // waves per GEMM block -> 128-row stripe

typedef __attribute__((ext_vector_type(16))) _Float16 v16h;
typedef __attribute__((ext_vector_type(8)))  _Float16 v8h;
typedef __attribute__((ext_vector_type(8)))  float    v8f;
typedef __attribute__((ext_vector_type(4)))  float    v4f;

// =======================================================================
// WMMA GEMM:  C[m,n] = sum_k A[m,k] * Bp[n,k]  (+ bias[n])
//   A  : [Mrows, K] f16 row-major   (K compile-time: 128 or 256)
//   Bp : [Ncols, K] f16 row-major   == B^T, i.e. C = A @ Bp^T
//   C  : [Mrows, ldc] f32
// Block: 256 threads = 8 waves. Block tile 128(rows) x 64(cols).
// The 64-col B panel (contiguous 64*K halves) is DMA'd into LDS with
// GLOBAL_LOAD_ASYNC_TO_LDS_B128 (ASYNCcnt) and shared by all 8 waves;
// each wave runs 4 WMMAs per K-step on one A fragment; the K loop is
// fully unrolled (compile-time K) for straight-line WMMA scheduling.
// Fragment layouts per CDNA5 ISA 7.12.2 (16-bit A 16x32, B 32x16, f32 C).
// =======================================================================
template <int K>
__global__ __launch_bounds__(256)
void wmma_gemm_f16(const _Float16* __restrict__ A,
                   const _Float16* __restrict__ Bp,
                   float* __restrict__ C,
                   const float* __restrict__ bias,
                   int ldc)
{
    __shared__ __align__(16) _Float16 Bs[TN * K];   // 16/32 KB of 320 KB LDS

    const int lane = threadIdx.x & 31;
    const int wave = threadIdx.x >> 5;
    const int m0    = (blockIdx.y * GWAVES + wave) * 16;
    const int nBase = blockIdx.x * TN;
    const int g  = lane >> 4;     // half-wave group
    const int r  = lane & 15;

    // ---- async DMA of the contiguous B panel into LDS (no VGPR bounce) ----
    {
        const _Float16* src = Bp + (size_t)nBase * K;
        // flat shared pointer: low 32 bits == LDS byte offset (ISA 10.2 aperture)
        const unsigned ldsBase = (unsigned)(uintptr_t)(&Bs[0]);
        constexpr int nvec = (TN * K) / 8;      // 16-byte units
#pragma unroll
        for (int v = 0; v < nvec / 256; ++v) {
            const int u = v * 256 + threadIdx.x;
            const unsigned long long ga =
                (unsigned long long)(uintptr_t)(src + (size_t)u * 8);
            const unsigned ld = ldsBase + (unsigned)u * 16u;
            asm volatile("global_load_async_to_lds_b128 %0, %1, off"
                         :: "v"(ld), "v"(ga) : "memory");
        }
        asm volatile("s_wait_asynccnt 0x0" ::: "memory");
    }
    __syncthreads();

    const _Float16* pa = A + (size_t)(m0 + r) * K;

    v8f acc[4] = {{}, {}, {}, {}};
#pragma unroll
    for (int kk = 0; kk < K; kk += 32) {
        // A 16x32 f16 frag: lane r (group g): halves 0..7 -> K=kk+8g+j,
        //                                     halves 8..15 -> K=kk+16+8g+(j-8)
        v8h a0 = *(const v8h*)(pa + kk + 8 * g);
        v8h a1 = *(const v8h*)(pa + kk + 16 + 8 * g);
        if (kk + 32 < K) __builtin_prefetch(pa + kk + 32, 0, 0);  // global_prefetch_b8
        v16h af;
#pragma unroll
        for (int j = 0; j < 8; ++j) { af[j] = a0[j]; af[j + 8] = a1[j]; }

        // preload all 4 B frags from LDS (8 x ds_load_b128, pipelined), ...
        v16h bf[4];
#pragma unroll
        for (int q = 0; q < 4; ++q) {
            const _Float16* pb = Bs + (size_t)(q * 16 + r) * K + kk + 8 * g;
            v8h b0 = *(const v8h*)(pb);
            v8h b1 = *(const v8h*)(pb + 16);
#pragma unroll
            for (int j = 0; j < 8; ++j) { bf[q][j] = b0[j]; bf[q][j + 8] = b1[j]; }
        }
        // ... then the back-to-back 4-WMMA chain on one A fragment
#pragma unroll
        for (int q = 0; q < 4; ++q) {
            acc[q] = __builtin_amdgcn_wmma_f32_16x16x32_f16(
                         false, af, false, bf[q], (short)0, acc[q], false, false);
        }
    }

    // ---- store: f32 C tile layout: reg rg holds (m0+rg+8g, n0+r)
#pragma unroll
    for (int q = 0; q < 4; ++q) {
        const int n = nBase + q * 16 + r;
        const float bv = bias ? bias[n] : 0.0f;
#pragma unroll
        for (int rg = 0; rg < 8; ++rg) {
            const int m = m0 + rg + 8 * g;
            C[(size_t)m * ldc + n] = acc[q][rg] + bv;
        }
    }
}

// ---------------- gather columns of pre_W + bias -> f16 x [N0, HD] -------
__global__ void gather_embed(const float* __restrict__ preW,   // [HD, IN_SIZE]
                             const float* __restrict__ preb,   // [HD]
                             const int*   __restrict__ nid,    // [N0]
                             _Float16*    __restrict__ xh)     // [N0, HD]
{
    const int idx = blockIdx.x * blockDim.x + threadIdx.x;   // N0*HD threads
    const int c = idx & (HD - 1);
    const int n = idx >> 8;
    const int id = nid[n];
    const float v = preW[(size_t)c * IN_SIZE + id] + preb[c];
    xh[idx] = (_Float16)v;
}

// ---------------- alpha[n,h] = <x_l[n, h*64 : h*64+64], att[h]> ----------
__global__ void alpha_kernel(const float* __restrict__ xl,   // [n, HD]
                             const float* __restrict__ att,  // [HEADS*DIMC]
                             float* __restrict__ alpha,      // [n, HEADS]
                             int nrows)
{
    const int idx = blockIdx.x * blockDim.x + threadIdx.x;
    if (idx >= nrows * HEADS) return;
    const int h = idx & 3;
    const int n = idx >> 2;
    const v4f* xr = (const v4f*)(xl + (size_t)n * HD + h * DIMC);
    const v4f* at = (const v4f*)(att + h * DIMC);
    float s = 0.f;
#pragma unroll
    for (int d = 0; d < DIMC / 4; ++d) {
        const v4f x = xr[d], a = at[d];
        s += x[0] * a[0] + x[1] * a[1] + x[2] * a[2] + x[3] * a[3];
    }
    alpha[idx] = s;
}

// ---------------- fixed-degree-16 edge softmax + aggregate ---------------
// e_dst is repeat(arange(ndst), 16): dst d owns edges [d*16, d*16+16).
// Block = 256 threads = 4 dst nodes; 64 threads/dst, 4 feats each (one head).
__global__ __launch_bounds__(256)
void gat_aggregate(const float* __restrict__ xl,     // [nsrc, HD]
                   const float* __restrict__ alpha,  // [nsrc, HEADS]
                   const int*   __restrict__ esrc,   // [ndst*DEG]
                   const float* __restrict__ bias,   // [HD]
                   float*       __restrict__ outF,   // [ndst, HD]
                   _Float16*    __restrict__ outH)   // [ndst, HD] or null
{
    const int dst = blockIdx.x * 4 + (threadIdx.x >> 6);
    const int t = threadIdx.x & 63;
    const int h = t >> 4;
    const int c = h * DIMC + (t & 15) * 4;

    int srcs[DEG];
    float lg[DEG];
    float mx = -1e30f;
#pragma unroll
    for (int j = 0; j < DEG; ++j) {
        const int s = esrc[dst * DEG + j];
        const float a = alpha[s * HEADS + h];
        const float l = a > 0.f ? a : NEG_SLOPE * a;
        srcs[j] = s; lg[j] = l;
        mx = fmaxf(mx, l);
    }
    float den = 0.f;
#pragma unroll
    for (int j = 0; j < DEG; ++j) { const float e = __expf(lg[j] - mx); lg[j] = e; den += e; }
    const float inv = 1.f / (den + 1e-16f);

    v4f acc = {};
#pragma unroll
    for (int j = 0; j < DEG; ++j) {
        const v4f x = *(const v4f*)(xl + (size_t)srcs[j] * HD + c);
        const float w = lg[j] * inv;
        acc[0] += w * x[0]; acc[1] += w * x[1]; acc[2] += w * x[2]; acc[3] += w * x[3];
    }
    const int base = dst * HD + c;
    if (outH) {
#pragma unroll
        for (int q = 0; q < 4; ++q) {
            const float o = acc[q] + bias[c + q];
            outF[base + q] = o;
            outH[base + q] = (_Float16)o;
        }
    } else {
#pragma unroll
        for (int q = 0; q < 4; ++q) outF[base + q] = acc[q] + bias[c + q];
    }
}

// ---------------- x_mod accumulate ---------------------------------------
__global__ void accum_kernel(const float* __restrict__ g0,   // [>=BATCH, HD]
                             const float* __restrict__ g1,   // [BATCH, HD]
                             const float* __restrict__ im,   // [BATCH, MMOD]
                             const float* __restrict__ scales, // [MMOD]
                             int mod, int first,
                             float* __restrict__ xmod)       // [BATCH, HD]
{
    const int idx = blockIdx.x * blockDim.x + threadIdx.x;   // BATCH*HD
    const int b = idx >> 8;
    const float s = scales[mod] * im[b * MMOD + mod];
    const float v = s * (2.f * g0[idx] + 2.f * g1[idx]);
    xmod[idx] = (first ? 0.f : xmod[idx]) + v;
}

// ---------------- f32 -> f16 ---------------------------------------------
__global__ void f32_to_f16(const float* __restrict__ in, _Float16* __restrict__ out, int n)
{
    const int idx = blockIdx.x * blockDim.x + threadIdx.x;
    if (idx < n) out[idx] = (_Float16)in[idx];
}

// ---------------- Threefry-2x32-20 (JAX PRNG, key(42) = (0, 42)) ---------
__device__ __forceinline__ unsigned rotl32(unsigned x, int d) { return (x << d) | (x >> (32 - d)); }

__device__ __forceinline__ void threefry2x32(unsigned x0, unsigned x1,
                                             unsigned* o0, unsigned* o1)
{
    const unsigned k0 = 0u, k1 = 42u;
    const unsigned ks[3] = { k0, k1, 0x1BD11BDAu ^ k0 ^ k1 };
    x0 += ks[0]; x1 += ks[1];
    const int RA[4] = {13, 15, 26, 6};
    const int RB[4] = {17, 29, 16, 24};
#pragma unroll
    for (int blk = 0; blk < 5; ++blk) {
        const int* R = (blk & 1) ? RB : RA;
#pragma unroll
        for (int i = 0; i < 4; ++i) { x0 += x1; x1 = rotl32(x1, R[i]); x1 ^= x0; }
        const unsigned gg = blk + 1;
        x0 += ks[gg % 3];
        x1 += ks[(gg + 1) % 3] + gg;
    }
    *o0 = x0; *o1 = x1;
}

// interp(): random 0/1 per (b, mod) via threefry, degenerate-row fix,
// masked uniform softmax -> im[b, mod]
__global__ void interp_kernel(const float* __restrict__ masks, float* __restrict__ im)
{
    const int b = blockIdx.x * blockDim.x + threadIdx.x;
    if (b >= BATCH) return;
    const int HALF = (BATCH * MMOD) / 2;   // jax random_bits counter split
    float r[MMOD];
#pragma unroll
    for (int k = 0; k < MMOD; ++k) {
        const int w = b * MMOD + k;
        const int j = (w < HALF) ? w : (w - HALF);
        unsigned o0, o1;
        threefry2x32((unsigned)j, (unsigned)(j + HALF), &o0, &o1);
        const unsigned bits = (w < HALF) ? o0 : o1;
        r[k] = (float)(bits & 1u);          // randint(0, 2)
    }
    if (r[0] + r[1] + r[2] == 0.f) { r[0] = r[1] = r[2] = 1.f; }  // floor(+1) fix
    float m[MMOD]; float cnt = 0.f;
#pragma unroll
    for (int k = 0; k < MMOD; ++k) {
        m[k] = (masks[b * MMOD + k] != 0.f) ? r[k] : 0.f;
        cnt += m[k];
    }
#pragma unroll
    for (int k = 0; k < MMOD; ++k)
        im[b * MMOD + k] = (cnt == 0.f) ? (1.f / MMOD) : (m[k] / cnt);
}

// scales = softmax(scales_param); write to ws + d_out tail
__global__ void scales_kernel(const float* __restrict__ p,
                              float* __restrict__ ws_scales,
                              float* __restrict__ out_tail)
{
    if (threadIdx.x == 0 && blockIdx.x == 0) {
        const float a0 = p[0], a1 = p[1], a2 = p[2];
        const float mx = fmaxf(a0, fmaxf(a1, a2));
        const float e0 = __expf(a0 - mx), e1 = __expf(a1 - mx), e2 = __expf(a2 - mx);
        const float inv = 1.f / (e0 + e1 + e2);
        const float s0 = e0 * inv, s1 = e1 * inv, s2 = e2 * inv;
        ws_scales[0] = s0; ws_scales[1] = s1; ws_scales[2] = s2;
        out_tail[0] = s0; out_tail[1] = s1; out_tail[2] = s2;
    }
}

// =======================================================================
extern "C" void kernel_launch(void* const* d_in, const int* in_sizes, int n_in,
                              void* d_out, int out_size, void* d_ws, size_t ws_size,
                              hipStream_t stream) {
    (void)in_sizes; (void)n_in; (void)out_size; (void)ws_size;

    const float* masks   = (const float*)d_in[0];
    const int*   n_ids   = (const int*)  d_in[1];   // [M, N0]
    const int*   e0_src  = (const int*)  d_in[2];   // [M, E0]
    const int*   e1_src  = (const int*)  d_in[4];   // [M, E1]
    const float* pre_W   = (const float*)d_in[6];   // [M, HD, IN_SIZE]
    const float* pre_b   = (const float*)d_in[7];   // [M, HD]
    const float* gat_W   = (const float*)d_in[8];   // [M, HD, HD]
    const float* gat_att = (const float*)d_in[9];   // [M, 1, H, D]
    const float* gat_b   = (const float*)d_in[10];  // [M, HD]
    const float* sc_par  = (const float*)d_in[11];  // [1, M]
    const float* emb_W   = (const float*)d_in[12];  // [128, HD]
    const float* emb_b   = (const float*)d_in[13];  // [128]

    float* out = (float*)d_out;
    const size_t DOT_SZ = (size_t)BATCH * BATCH;
    const size_t EMB_SZ = (size_t)BATCH * 128;
    float* out_emb  = out + DOT_SZ;
    float* out_tail = out + DOT_SZ + EMB_SZ;

    // ---- workspace carve-up ----
    char* ws = (char*)d_ws;
    size_t off = 0;
    auto alloc = [&](size_t bytes) -> void* {
        off = (off + 255) & ~(size_t)255;
        void* p = ws + off;
        off += bytes;
        return p;
    };
    _Float16* gatW16 = (_Float16*)alloc((size_t)MMOD * HD * HD * 2);
    _Float16* embW16 = (_Float16*)alloc((size_t)128 * HD * 2);
    _Float16* xh     = (_Float16*)alloc((size_t)N0S * HD * 2);
    float*    xl     = (float*)   alloc((size_t)N0S * HD * 4);
    float*    alphaB = (float*)   alloc((size_t)N0S * HEADS * 4);
    float*    g0f    = (float*)   alloc((size_t)N1S * HD * 4);
    _Float16* g0h    = (_Float16*)alloc((size_t)N1S * HD * 2);
    float*    g1f    = (float*)   alloc((size_t)BATCH * HD * 4);
    float*    xmod   = (float*)   alloc((size_t)BATCH * HD * 4);
    _Float16* xmodh  = (_Float16*)alloc((size_t)BATCH * HD * 2);
    _Float16* embh   = (_Float16*)alloc((size_t)BATCH * 128 * 2);
    float*    im     = (float*)   alloc((size_t)BATCH * MMOD * 4);
    float*    wsSc   = (float*)   alloc(16);

    // ---- small precomputations ----
    f32_to_f16<<<(MMOD * HD * HD + 255) / 256, 256, 0, stream>>>(gat_W, gatW16, MMOD * HD * HD);
    f32_to_f16<<<(128 * HD + 255) / 256, 256, 0, stream>>>(emb_W, embW16, 128 * HD);
    interp_kernel<<<BATCH / 256, 256, 0, stream>>>(masks, im);
    scales_kernel<<<1, 32, 0, stream>>>(sc_par, wsSc, out_tail);

    // ---- per-modality pipeline ----
    for (int i = 0; i < MMOD; ++i) {
        const _Float16* Wg = gatW16 + (size_t)i * HD * HD;
        const float* att_i = gat_att + (size_t)i * HEADS * DIMC;
        const float* gb_i  = gat_b + (size_t)i * HD;

        // x = gather(pre_W) + pre_b  -> f16 [N0, HD]
        gather_embed<<<(size_t)N0S * HD / 256, 256, 0, stream>>>(
            pre_W + (size_t)i * HD * IN_SIZE, pre_b + (size_t)i * HD,
            n_ids + (size_t)i * N0S, xh);

        // x_l = x @ W^T  (f32)
        wmma_gemm_f16<HD><<<dim3(HD / TN, N0S / 128), 256, 0, stream>>>(
            xh, Wg, xl, nullptr, HD);

        // alpha[n,h]
        alpha_kernel<<<(N0S * HEADS) / 256, 256, 0, stream>>>(xl, att_i, alphaB, N0S);

        // layer-0 aggregate -> g0 [N1, HD] (f32 + f16)
        gat_aggregate<<<N1S / 4, 256, 0, stream>>>(xl, alphaB, e0_src + (size_t)i * E0S,
                                                   gb_i, g0f, g0h);

        // x_l2 = g0 @ W^T
        wmma_gemm_f16<HD><<<dim3(HD / TN, N1S / 128), 256, 0, stream>>>(
            g0h, Wg, xl, nullptr, HD);

        alpha_kernel<<<(N1S * HEADS) / 256, 256, 0, stream>>>(xl, att_i, alphaB, N1S);

        // layer-1 aggregate -> g1 [B, HD]
        gat_aggregate<<<BATCH / 4, 256, 0, stream>>>(xl, alphaB, e1_src + (size_t)i * E1S,
                                                     gb_i, g1f, nullptr);

        // x_mod += scales[i]*im[:,i]*(2*g0[:B] + 2*g1)
        accum_kernel<<<(BATCH * HD) / 256, 256, 0, stream>>>(
            g0f, g1f, im, wsSc, i, (i == 0) ? 1 : 0, xmod);
    }

    // ---- emb = x_mod @ emb_W^T + emb_b  -> out_emb [B, 128] ----
    f32_to_f16<<<(BATCH * HD) / 256, 256, 0, stream>>>(xmod, xmodh, BATCH * HD);
    wmma_gemm_f16<HD><<<dim3(128 / TN, BATCH / 128), 256, 0, stream>>>(
        xmodh, embW16, out_emb, emb_b, 128);

    // ---- dot = emb @ emb^T -> out [B, B] ----
    f32_to_f16<<<(BATCH * 128) / 256, 256, 0, stream>>>(out_emb, embh, BATCH * 128);
    wmma_gemm_f16<128><<<dim3(BATCH / TN, BATCH / 128), 256, 0, stream>>>(
        embh, embh, out, nullptr, BATCH);
}
